// ViTBlockworldModel_46806553592106
// MI455X (gfx1250) — compile-verified, matrix operand order checked by reading, not run on hardware
//
#include <hip/hip_runtime.h>
#include <hip/hip_bf16.h>
#include <math.h>

// ---------------------------------------------------------------------------
// ViT block world-model for MI455X (gfx1250, wave32, WMMA).
// Matmuls: v_wmma_f32_16x16x32_bf16, fp32 tiles DMA'd to LDS via
// global_load_async_to_lds_b128 (double buffered), bf16 conversion at
// fragment build. Attention: flash-style WMMA kernel.
// ---------------------------------------------------------------------------

typedef __attribute__((ext_vector_type(16))) __bf16 v16bf;
typedef __attribute__((ext_vector_type(8)))  __bf16 v8bf;
typedef __attribute__((ext_vector_type(8)))  float  v8f;
typedef __attribute__((ext_vector_type(4)))  int    v4i;

#define WMMA_BF16(a, b, c) \
  __builtin_amdgcn_wmma_f32_16x16x32_bf16(false, (a), false, (b), (short)0, (c), false, false)

// ---- async global->LDS copy (ASYNCcnt path), with synchronous fallback ----
#if __has_builtin(__builtin_amdgcn_global_load_async_to_lds_b128)
typedef __attribute__((address_space(1))) v4i* as1_v4i_ptr;  // global int4*
typedef __attribute__((address_space(3))) v4i* as3_v4i_ptr;  // LDS int4*
__device__ __forceinline__ void async_copy16(const float* g, float* l) {
  __builtin_amdgcn_global_load_async_to_lds_b128(
      (as1_v4i_ptr)(void*)g, (as3_v4i_ptr)(void*)l, 0, 0);
}
__device__ __forceinline__ void async_wait0() {
#if __has_builtin(__builtin_amdgcn_s_wait_asynccnt)
  __builtin_amdgcn_s_wait_asynccnt(0);
#else
  asm volatile("s_wait_asynccnt 0" ::: "memory");
#endif
}
#else
__device__ __forceinline__ void async_copy16(const float* g, float* l) {
  *(float4*)l = *(const float4*)g;  // synchronous fallback
}
__device__ __forceinline__ void async_wait0() {}
#endif

// Build a 16x32 bf16 fragment row from a K-contiguous fp32 LDS row.
// Per CDNA5 ISA 16-bit A layout: lanes 0-15 hold K {0..7, 16..23},
// lanes 16-31 hold K {8..15, 24..31} -> two 8-elem chunks at hl*8, 16+hl*8.
__device__ __forceinline__ v16bf load_frag_f32(const float* rowp, int hl) {
  union { v16bf v; __bf16 e[16]; } u;
#pragma unroll
  for (int i = 0; i < 8; ++i) {
    u.e[i]     = (__bf16)rowp[hl * 8 + i];
    u.e[8 + i] = (__bf16)rowp[16 + hl * 8 + i];
  }
  return u.v;
}

// bf16-LDS variant (used by flash attention).
__device__ __forceinline__ v16bf load_frag(const __bf16* rowp, int hl) {
  union { v16bf v; v8bf h[2]; } u;
  u.h[0] = *(const v8bf*)(rowp + hl * 8);
  u.h[1] = *(const v8bf*)(rowp + 16 + hl * 8);
  return u.v;
}

enum { EPI_BIAS = 0, EPI_GELU = 1, EPI_RES = 2, EPI_POSIMG = 3 };

// ---------------------------------------------------------------------------
// GEMM:  C = epilogue( A(MxK,f32) * W(NxK,f32)^T )
// Block: 128 threads = 4 waves; block tile 64(M) x 128(N); K-step 32.
// Wave tile 32x64 = 2x4 WMMA accumulators (8 WMMAs per wave per K-step).
// LDS tiles kept fp32 and double buffered, filled by async DMA.
// crow = row + crow_off + ((row >> crow_shift) * crow_add)   (write remap)
// ---------------------------------------------------------------------------
__global__ __launch_bounds__(128) void gemm_bf16_kernel(
    const float* __restrict__ A, int lda,
    const float* __restrict__ W,                 // N x K row-major
    const float* __restrict__ bias,              // [N] or null
    const float* __restrict__ aux, int aux_mask, // aux[(row&aux_mask)*N+col]
    const float* __restrict__ res, int ldres,
    float* __restrict__ C, int ldc,
    int M, int N, int K,
    int crow_off, int crow_shift, int crow_add,
    int epi) {
  __shared__ __align__(16) float Asf[2][64 * 32];
  __shared__ __align__(16) float Bsf[2][128 * 32];

  const int tid  = threadIdx.x;
  const int wid  = tid >> 5;
  const int lane = tid & 31;
  const int hl   = lane >> 4;
  const int l16  = lane & 15;
  const int blockM = blockIdx.y * 64;
  const int blockN = blockIdx.x * 128;
  const int wm = (wid >> 1) * 32;   // wave M offset 0/32
  const int wn = (wid & 1) * 64;    // wave N offset 0/64

  const int trow = tid >> 3;         // 0..15
  const int tcol = (tid & 7) * 4;    // 0..28, 16B chunks

  v8f acc[2][4] = {};

  const int nk = K >> 5;

#define STAGE(bufi, k0)                                                        \
  do {                                                                         \
    _Pragma("unroll")                                                          \
    for (int i = 0; i < 4; ++i)                                                \
      async_copy16(A + (size_t)(blockM + trow + i * 16) * lda + (k0) + tcol,   \
                   &Asf[bufi][(trow + i * 16) * 32 + tcol]);                   \
    _Pragma("unroll")                                                          \
    for (int i = 0; i < 8; ++i)                                                \
      async_copy16(W + (size_t)(blockN + trow + i * 16) * K + (k0) + tcol,     \
                   &Bsf[bufi][(trow + i * 16) * 32 + tcol]);                   \
  } while (0)

  STAGE(0, 0);
  async_wait0();
  __syncthreads();

  for (int kt = 0; kt < nk; ++kt) {
    const int cur = kt & 1;
    if (kt + 1 < nk) STAGE(cur ^ 1, (kt + 1) * 32);

    v16bf af[2], bfg[4];
#pragma unroll
    for (int i = 0; i < 2; ++i) af[i]  = load_frag_f32(&Asf[cur][(wm + i * 16 + l16) * 32], hl);
#pragma unroll
    for (int j = 0; j < 4; ++j) bfg[j] = load_frag_f32(&Bsf[cur][(wn + j * 16 + l16) * 32], hl);
#pragma unroll
    for (int i = 0; i < 2; ++i)
#pragma unroll
      for (int j = 0; j < 4; ++j) acc[i][j] = WMMA_BF16(af[i], bfg[j], acc[i][j]);

    if (kt + 1 < nk) async_wait0();
    __syncthreads();
  }
#undef STAGE

  // Epilogue. C layout: slot r -> row = r + hl*8 (within 16x16), col = l16.
#pragma unroll
  for (int i = 0; i < 2; ++i) {
#pragma unroll
    for (int j = 0; j < 4; ++j) {
#pragma unroll
      for (int r = 0; r < 8; ++r) {
        int row = blockM + wm + i * 16 + r + hl * 8;
        int col = blockN + wn + j * 16 + l16;
        float v = acc[i][j][r];
        if (bias) v += bias[col];
        if (epi == EPI_POSIMG) v += aux[(size_t)(row & aux_mask) * N + col];
        if (epi == EPI_GELU)   v = 0.5f * v * (1.0f + erff(v * 0.70710678118f));
        int crow = row + crow_off + ((row >> crow_shift) * crow_add);
        if (epi == EPI_RES)    v += res[(size_t)crow * ldres + col];
        C[(size_t)crow * ldc + col] = v;
      }
    }
  }
}

// ---------------------------------------------------------------------------
// Flash attention, all-WMMA. Block = 128 threads (4 waves), one (b,h) and a
// 64-row Q chunk per block. Each wave owns a 16-row strip. S=512, dh=64.
// ---------------------------------------------------------------------------
__global__ __launch_bounds__(128) void flash_attn_kernel(
    const float* __restrict__ qkv,   // (B*S, 3D)
    float* __restrict__ o,           // (B*S, D)
    int S, int D3, int D, int nh) {
  const int dh = 64;
  const int bh = blockIdx.y;
  const int b = bh / nh, h = bh % nh;
  const int qbase = blockIdx.x * 64;
  const int tid = threadIdx.x;
  const int wid = tid >> 5, lane = tid & 31;
  const int hl = lane >> 4, l16 = lane & 15;

  __shared__ __align__(16) __bf16 Kl[64 * 64];  // keys x dh
  __shared__ __align__(16) __bf16 Vt[64 * 64];  // dh x keys (transposed V)
  __shared__ __align__(16) __bf16 Pl[64 * 64];  // probs, wave-private strips

  // Q fragments for this wave's 16 rows (K-dim = dh = 64 -> 2 frags).
  v16bf qf[2];
  {
    const int qrow = qbase + wid * 16 + l16;
    const float* qp = qkv + (size_t)(b * S + qrow) * D3 + h * dh;
#pragma unroll
    for (int f = 0; f < 2; ++f) {
      union { v16bf v; __bf16 e[16]; } u;
#pragma unroll
      for (int i = 0; i < 8; ++i) {
        u.e[i]     = (__bf16)qp[f * 32 + hl * 8 + i];
        u.e[8 + i] = (__bf16)qp[f * 32 + 16 + hl * 8 + i];
      }
      qf[f] = u.v;
    }
  }

  float mrow[8], lrow[8];
  v8f accO[4] = {};
#pragma unroll
  for (int r = 0; r < 8; ++r) { mrow[r] = -1e30f; lrow[r] = 0.0f; }
  const float scale = 0.125f;  // 1/sqrt(64)

  for (int kc = 0; kc < S; kc += 64) {
    {  // stage K chunk and transposed V chunk
      const int key = tid >> 1, dc0 = (tid & 1) * 32;
      const float* kp = qkv + (size_t)(b * S + kc + key) * D3 + D + h * dh + dc0;
      const float* vp = qkv + (size_t)(b * S + kc + key) * D3 + 2 * D + h * dh + dc0;
      if (kc + 64 < S) {  // prefetch next chunk (global_prefetch_b8)
        __builtin_prefetch(kp + (size_t)64 * D3, 0, 1);
        __builtin_prefetch(vp + (size_t)64 * D3, 0, 1);
      }
#pragma unroll
      for (int i = 0; i < 32; ++i) {
        Kl[key * 64 + dc0 + i]   = (__bf16)kp[i];
        Vt[(dc0 + i) * 64 + key] = (__bf16)vp[i];
      }
    }
    __syncthreads();

    // scores = Q Kt : 16 x 64, 4 key-subtiles x 2 K-steps
    v8f accS[4] = {};
#pragma unroll
    for (int j = 0; j < 4; ++j)
#pragma unroll
      for (int kk = 0; kk < 2; ++kk) {
        v16bf kf = load_frag(&Kl[(j * 16 + l16) * 64 + kk * 32], hl);
        accS[j] = WMMA_BF16(qf[kk], kf, accS[j]);
      }

    // online softmax per row slot (rows r / r+8 in lane halves)
#pragma unroll
    for (int r = 0; r < 8; ++r) {
      float mx = -1e30f;
#pragma unroll
      for (int j = 0; j < 4; ++j) mx = fmaxf(mx, accS[j][r]);
#pragma unroll
      for (int off = 1; off < 16; off <<= 1) mx = fmaxf(mx, __shfl_xor(mx, off, 16));
      mx *= scale;
      float mnew = fmaxf(mrow[r], mx);
      float alpha = __expf(mrow[r] - mnew);
      mrow[r] = mnew;
      lrow[r] *= alpha;
#pragma unroll
      for (int j = 0; j < 4; ++j) accO[j][r] *= alpha;
      float rsum = 0.0f;
#pragma unroll
      for (int j = 0; j < 4; ++j) {
        float p = __expf(accS[j][r] * scale - mnew);
        rsum += p;
        Pl[(wid * 16 + r + hl * 8) * 64 + j * 16 + l16] = (__bf16)p;
      }
#pragma unroll
      for (int off = 1; off < 16; off <<= 1) rsum += __shfl_xor(rsum, off, 16);
      lrow[r] += rsum;
    }

    // accO += P V  (K-dim = 64 keys; Pl strip is wave-private)
#pragma unroll
    for (int kk = 0; kk < 2; ++kk) {
      v16bf pf = load_frag(&Pl[(wid * 16 + l16) * 64 + kk * 32], hl);
#pragma unroll
      for (int j = 0; j < 4; ++j) {
        v16bf vf = load_frag(&Vt[(j * 16 + l16) * 64 + kk * 32], hl);
        accO[j] = WMMA_BF16(pf, vf, accO[j]);
      }
    }
    __syncthreads();
  }

#pragma unroll
  for (int j = 0; j < 4; ++j)
#pragma unroll
    for (int r = 0; r < 8; ++r) {
      int row = qbase + wid * 16 + r + hl * 8;
      int col = h * dh + j * 16 + l16;
      o[(size_t)(b * S + row) * D + col] = accO[j][r] / lrow[r];
    }
}

// ---------------------------------------------------------------------------
// LayerNorm: one block (256 thr) per row of D=768.
// ---------------------------------------------------------------------------
__global__ __launch_bounds__(256) void layernorm_kernel(
    const float* __restrict__ x, const float* __restrict__ g,
    const float* __restrict__ bta, float* __restrict__ y, int D) {
  const int row = blockIdx.x;
  const float* xr = x + (size_t)row * D;
  const int tid = threadIdx.x;
  float s = 0.0f, s2 = 0.0f;
  for (int i = tid; i < D; i += 256) { float v = xr[i]; s += v; s2 += v * v; }
#pragma unroll
  for (int off = 1; off < 32; off <<= 1) {
    s += __shfl_xor(s, off, 32);
    s2 += __shfl_xor(s2, off, 32);
  }
  __shared__ float rs[8], rs2[8];
  if ((tid & 31) == 0) { rs[tid >> 5] = s; rs2[tid >> 5] = s2; }
  __syncthreads();
  float ts = 0.0f, ts2 = 0.0f;
#pragma unroll
  for (int i = 0; i < 8; ++i) { ts += rs[i]; ts2 += rs2[i]; }
  float mean = ts / (float)D;
  float var = ts2 / (float)D - mean * mean;
  float rstd = rsqrtf(var + 1e-5f);
  for (int i = tid; i < D; i += 256)
    y[(size_t)row * D + i] = (xr[i] - mean) * rstd * g[i] + bta[i];
}

// sincos2d positional table: one block per token.
__global__ void pos_kernel(float* __restrict__ pos, const int* __restrict__ idx,
                           int Wdim, int D) {
  int t = blockIdx.x;
  int p = idx ? idx[t] : t;
  float y = (float)(p / Wdim);
  float x = (float)(p % Wdim);
  int half = D / 2;  // 384
  for (int i = threadIdx.x; i < half / 2; i += blockDim.x) {
    float div = __expf((float)(2 * i) * (-9.210340371976184f / (float)half));
    pos[(size_t)t * D + 2 * i]            = __sinf(y * div);
    pos[(size_t)t * D + 2 * i + 1]        = __cosf(y * div);
    pos[(size_t)t * D + half + 2 * i]     = __sinf(x * div);
    pos[(size_t)t * D + half + 2 * i + 1] = __cosf(x * div);
  }
}

// Gather h_cv tokens and write joint[:, :Ncv] = h_cv + pos.
__global__ void embed_cv_kernel(const float* __restrict__ h_map,
                                const int* __restrict__ idx,
                                const float* __restrict__ pos_cv,
                                float* __restrict__ joint, float* __restrict__ h_cv,
                                int Ncv, int S, int HW, int D) {
  int r = blockIdx.x;  // b*Ncv + t
  int b = r / Ncv, t = r % Ncv;
  const float* src = h_map + ((size_t)b * HW + idx[t]) * D;
  float* jp = joint + ((size_t)b * S + t) * D;
  float* hp = h_cv + (size_t)r * D;
  for (int d = threadIdx.x; d < D; d += blockDim.x) {
    float v = src[d];
    hp[d] = v;
    jp[d] = v + pos_cv[(size_t)t * D + d];
  }
}

// Im2col: frame (B,3,256,256) -> patches (B*256, 768), col = ch*256+py*16+px.
__global__ void patch_extract_kernel(const float* __restrict__ frame,
                                     float* __restrict__ patches) {
  int r = blockIdx.x;  // b*256 + g
  int b = r >> 8, g = r & 255;
  int gy = g >> 4, gx = g & 15;
  for (int c = threadIdx.x; c < 768; c += blockDim.x) {
    int ch = c >> 8, w = c & 255;
    int py = w >> 4, px = w & 15;
    patches[(size_t)r * 768 + c] =
        frame[(((size_t)b * 3 + ch) * 256 + gy * 16 + py) * 256 + gx * 16 + px];
  }
}

// gateIn = concat(h_cv, h_new) per cv token.
__global__ void gate_input_kernel(const float* __restrict__ h_cv,
                                  const float* __restrict__ xn,
                                  float* __restrict__ gateIn, int Ncv, int S, int D) {
  int r = blockIdx.x;
  int b = r / Ncv, t = r % Ncv;
  for (int d = threadIdx.x; d < D; d += blockDim.x) {
    gateIn[(size_t)r * 2 * D + d]     = h_cv[(size_t)r * D + d];
    gateIn[(size_t)r * 2 * D + D + d] = xn[((size_t)b * S + t) * D + d];
  }
}

__global__ void copy4_kernel(const float4* __restrict__ src, float4* __restrict__ dst,
                             size_t n4) {
  size_t i = (size_t)blockIdx.x * blockDim.x + threadIdx.x;
  if (i < n4) dst[i] = src[i];
}

// out[idx tokens] = (1-sigmoid(glog))*h_cv + sigmoid(glog)*h_new
__global__ void gate_scatter_kernel(const float* __restrict__ glog,
                                    const float* __restrict__ h_cv,
                                    const float* __restrict__ xn,
                                    const int* __restrict__ idx,
                                    float* __restrict__ out,
                                    int Ncv, int S, int HW, int D) {
  int r = blockIdx.x;
  int b = r / Ncv, t = r % Ncv;
  float* op = out + ((size_t)b * HW + idx[t]) * D;
  for (int d = threadIdx.x; d < D; d += blockDim.x) {
    float g = 1.0f / (1.0f + __expf(-glog[(size_t)r * D + d]));
    float hc = h_cv[(size_t)r * D + d];
    float hn = xn[((size_t)b * S + t) * D + d];
    op[d] = (1.0f - g) * hc + g * hn;
  }
}

// ---------------------------------------------------------------------------
extern "C" void kernel_launch(void* const* d_in, const int* in_sizes, int n_in,
                              void* d_out, int out_size, void* d_ws, size_t ws_size,
                              hipStream_t stream) {
  const int B = 32, D = 768, depth = 6, dff = 3072, Ncv = 256, S = 512;
  const int nh = 12, HW = 33 * 33, D3 = 3 * D;
  const int M = B * S;       // 16384
  const int Mcv = B * Ncv;   // 8192

  const float* h_map  = (const float*)d_in[0];
  const float* frame  = (const float*)d_in[1];
  const float* patch_w = (const float*)d_in[2];
  const float* patch_b = (const float*)d_in[3];
  const float* ln1_g  = (const float*)d_in[4];
  const float* ln1_b  = (const float*)d_in[5];
  const float* wqkv   = (const float*)d_in[6];
  const float* bqkv   = (const float*)d_in[7];
  const float* wo     = (const float*)d_in[8];
  const float* bo     = (const float*)d_in[9];
  const float* ln2_g  = (const float*)d_in[10];
  const float* ln2_b  = (const float*)d_in[11];
  const float* w1     = (const float*)d_in[12];
  const float* b1     = (const float*)d_in[13];
  const float* w2     = (const float*)d_in[14];
  const float* b2     = (const float*)d_in[15];
  const float* norm_g = (const float*)d_in[16];
  const float* norm_b = (const float*)d_in[17];
  const float* gate_w = (const float*)d_in[18];
  const float* gate_b = (const float*)d_in[19];
  const int*   idx    = (const int*)d_in[20];
  float* out = (float*)d_out;

  // workspace layout (fp32)
  float* ws = (float*)d_ws;
  float* pos_cv  = ws; ws += (size_t)Ncv * D;
  float* pos_img = ws; ws += (size_t)256 * D;
  float* h_cv    = ws; ws += (size_t)Mcv * D;
  float* joint   = ws; ws += (size_t)M * D;
  float* xn      = ws; ws += (size_t)M * D;
  float* qkvb    = ws; ws += (size_t)M * D3;
  float* obuf    = ws; ws += (size_t)M * D;
  float* hidden  = ws; ws += (size_t)M * dff;
  float* patches = ws; ws += (size_t)Mcv * D;
  float* gateIn  = ws; ws += (size_t)Mcv * 2 * D;
  float* glog    = ws; ws += (size_t)Mcv * D;

  const int NOSHIFT = 30;  // (row >> 30) == 0 for all rows

  // positional tables + token/patch prep
  pos_kernel<<<Ncv, 256, 0, stream>>>(pos_cv, idx, 33, D);
  pos_kernel<<<256, 256, 0, stream>>>(pos_img, nullptr, 16, D);
  embed_cv_kernel<<<Mcv, 256, 0, stream>>>(h_map, idx, pos_cv, joint, h_cv, Ncv, S, HW, D);
  patch_extract_kernel<<<Mcv, 256, 0, stream>>>(frame, patches);

  // patch embed GEMM -> joint image rows (row remap r -> b*512 + 256 + g)
  gemm_bf16_kernel<<<dim3(D / 128, Mcv / 64), 128, 0, stream>>>(
      patches, D, patch_w, patch_b, pos_img, 255, nullptr, 0, joint, D,
      Mcv, D, D, 256, 8, 256, EPI_POSIMG);

  for (int i = 0; i < depth; ++i) {
    const float* wqkv_i = wqkv + (size_t)i * D3 * D;
    const float* bqkv_i = bqkv + (size_t)i * D3;
    const float* wo_i   = wo + (size_t)i * D * D;
    const float* bo_i   = bo + (size_t)i * D;
    const float* w1_i   = w1 + (size_t)i * dff * D;
    const float* b1_i   = b1 + (size_t)i * dff;
    const float* w2_i   = w2 + (size_t)i * D * dff;
    const float* b2_i   = b2 + (size_t)i * D;

    layernorm_kernel<<<M, 256, 0, stream>>>(joint, ln1_g + (size_t)i * D,
                                            ln1_b + (size_t)i * D, xn, D);
    gemm_bf16_kernel<<<dim3(D3 / 128, M / 64), 128, 0, stream>>>(
        xn, D, wqkv_i, bqkv_i, nullptr, 0, nullptr, 1, qkvb, D3,
        M, D3, D, 0, NOSHIFT, 0, EPI_BIAS);
    flash_attn_kernel<<<dim3(S / 64, B * nh), 128, 0, stream>>>(qkvb, obuf, S, D3, D, nh);
    gemm_bf16_kernel<<<dim3(D / 128, M / 64), 128, 0, stream>>>(
        obuf, D, wo_i, bo_i, nullptr, 0, joint, D, joint, D,
        M, D, D, 0, NOSHIFT, 0, EPI_RES);
    layernorm_kernel<<<M, 256, 0, stream>>>(joint, ln2_g + (size_t)i * D,
                                            ln2_b + (size_t)i * D, xn, D);
    gemm_bf16_kernel<<<dim3(dff / 128, M / 64), 128, 0, stream>>>(
        xn, D, w1_i, b1_i, nullptr, 0, nullptr, 1, hidden, dff,
        M, dff, D, 0, NOSHIFT, 0, EPI_GELU);
    gemm_bf16_kernel<<<dim3(D / 128, M / 64), 128, 0, stream>>>(
        hidden, dff, w2_i, b2_i, nullptr, 0, joint, D, joint, D,
        M, D, dff, 0, NOSHIFT, 0, EPI_RES);
  }

  layernorm_kernel<<<M, 256, 0, stream>>>(joint, norm_g, norm_b, xn, D);
  gate_input_kernel<<<Mcv, 256, 0, stream>>>(h_cv, xn, gateIn, Ncv, S, D);
  gemm_bf16_kernel<<<dim3(D / 128, Mcv / 64), 128, 0, stream>>>(
      gateIn, 2 * D, gate_w, gate_b, nullptr, 0, nullptr, 1, glog, D,
      Mcv, D, 2 * D, 0, NOSHIFT, 0, EPI_BIAS);

  size_t n4 = (size_t)B * HW * D / 4;
  copy4_kernel<<<(unsigned)((n4 + 255) / 256), 256, 0, stream>>>(
      (const float4*)h_map, (float4*)out, n4);
  gate_scatter_kernel<<<Mcv, 256, 0, stream>>>(glog, h_cv, xn, idx, out, Ncv, S, HW, D);
}
